// axs_89807766159734
// MI455X (gfx1250) — matrix operand order repaired and checked by reading, not compiled, not run on hardware
//
#include <hip/hip_runtime.h>
#include <hip/hip_bf16.h>

// ---------------------------------------------------------------------------
// out[b, p] = relu(weight[p]) * sum_{taps} exp(-0.5*||q2d - pos2d[p]||^2) * x[b, q]
// Recast as GEMM: out(32768x784) = X(32768x784) . W^T(784x784), f16 WMMA, f32 acc.
// 224-thread blocks (7 waves), M-block = 32: each wave owns 2 M-tiles x 7 N-tiles
// -> 14 unconditional v_wmma per K-step, each B fragment reused twice.
// X tiles are staged global->LDS with CDNA5 async copies (ASYNCcnt), so the
// fill never touches VGPRs / loadcnt and is fully hidden behind the WMMAs.
// ---------------------------------------------------------------------------

typedef __attribute__((ext_vector_type(16))) _Float16 v16h;
typedef __attribute__((ext_vector_type(16))) float    v16f;
typedef __attribute__((ext_vector_type(8)))  float    v8f;
typedef __attribute__((ext_vector_type(4)))  float    v4f;

#define H 28
#define P 784            // outputs = inputs = 28*28
#define NT 49            // 784/16 N-tiles
#define KT 25            // 800/32 K-tiles (K padded 784 -> 800)
#define TILE_HALFS 512   // one 32x16 B-fragment tile: 32 lanes x 16 halfs
#define MBLK 32          // batch rows per workgroup

// ---------------------------------------------------------------------------
// Kernel 1: build W pre-swizzled into the v_wmma B-matrix per-lane layout.
// Tile t = nt*25 + kt is 1024 bytes; lane l holds 16 contiguous halfs =
// column n = nt*16 + l%16, rows k = kt*32 + (l/16)*16 + 0..15.
// ---------------------------------------------------------------------------
__global__ void axs_build_w(const float* __restrict__ pos2d,
                            const float* __restrict__ weight,
                            _Float16* __restrict__ wswz) {
    int gid = blockIdx.x * blockDim.x + threadIdx.x;
    if (gid >= NT * KT * TILE_HALFS) return;

    int t    = gid / TILE_HALFS;       // tile index
    int u    = gid % TILE_HALFS;
    int lane = u >> 4;                 // 0..31
    int h    = u & 15;                 // element within lane
    int nt   = t / KT;
    int kt   = t % KT;
    int n    = nt * 16 + (lane & 15);            // output pixel index (< 784)
    int k    = kt * 32 + ((lane >> 4) << 4) + h; // input pixel index  (< 800)

    float val = 0.0f;
    if (k < P) {
        int oi = n / H, oj = n % H;    // output pixel 2d
        int ki = k / H, kj = k % H;    // input pixel 2d (in-bounds by construction)
        float px = pos2d[(oi * H + oj) * 2 + 0];
        float py = pos2d[(oi * H + oj) * 2 + 1];
        float ce0 = rintf(px);         // round-half-even == jnp.round
        float ce1 = rintf(py);
        float d0 = (float)ki - ce0;    // exact integer-valued float
        float d1 = (float)kj - ce1;
        if (fabsf(d0) <= 2.0f && fabsf(d1) <= 2.0f) {
            float ex = (float)ki - px;
            float ey = (float)kj - py;
            float w  = __expf(-0.5f * (ex * ex + ey * ey));
            float sw = weight[oi * H + oj];
            sw = sw > 0.0f ? sw : 0.0f;
            val = w * sw;
        }
    }
    wswz[t * TILE_HALFS + lane * 16 + h] = (_Float16)val;
}

// async global->LDS copy of 16 bytes (per active lane), ASYNCcnt-tracked
__device__ __forceinline__ void async_b128(uint32_t lds_off, const float* gaddr) {
    asm volatile("global_load_async_to_lds_b128 %0, %1, off"
                 :: "v"(lds_off), "v"(gaddr) : "memory");
}
__device__ __forceinline__ void wait_async0() {
    asm volatile("s_wait_asynccnt 0x0" ::: "memory");
}

// ---------------------------------------------------------------------------
// Kernel 2: GEMM. One 224-thread block (7 waves) per 32-batch row block.
// Per K-step: s_wait_asynccnt 0 -> barrier -> issue async fill of tile kt+1
// (fire-and-forget) -> ds_load + cvt two A fragments -> 7 B fragments ->
// 14 WMMA.  The async fill has a whole WMMA block + barrier to complete.
// ---------------------------------------------------------------------------
__global__ __launch_bounds__(224) void axs_gemm(const float* __restrict__ x,
                                                const _Float16* __restrict__ wswz,
                                                float* __restrict__ out) {
    __shared__ __align__(16) float tile[2][MBLK * 32];   // staged f32 X tiles

    const int tid  = threadIdx.x;
    const int wave = tid >> 5;           // 0..6
    const int lane = tid & 31;
    const int b0   = blockIdx.x * MBLK;  // batch row block

    // fill geometry: 128 threads x 8 contiguous f32 = 2 async b128 each
    const int fe = (tid & 127) << 3;     // 0..1016, step 8 (floats)
    const int fr = fe >> 5;              // row 0..31
    const int fc = fe & 31;              // col 0,8,16,24
    const bool filler = (tid < 128);
    const float* xrow = x + (size_t)(b0 + fr) * P + fc;

    v8f acc0[7], acc1[7];
#pragma unroll
    for (int f = 0; f < 7; ++f) {
        acc0[f] = (v8f){0.f,0.f,0.f,0.f,0.f,0.f,0.f,0.f};
        acc1[f] = (v8f){0.f,0.f,0.f,0.f,0.f,0.f,0.f,0.f};
    }

    // A-fragment LDS offsets (contiguous k order, as in the W tiles), in f32 elements:
    // lane l<16 : row l, K = 0..15 ; lane l+16 : K = 16..31
    const int aoff = (lane & 15) * 32 + ((lane >> 4) << 4);
    const _Float16* wbase = wswz + (size_t)(wave * 7) * KT * TILE_HALFS + lane * 16;

    // prologue: async-stage K-tile 0 into buffer 0 (kt=0 always in bounds)
    if (filler) {
        uint32_t lds = (uint32_t)(uintptr_t)(&tile[0][fe]);
        async_b128(lds,      xrow);
        async_b128(lds + 16, xrow + 4);
    }

    for (int kt = 0; kt < KT; ++kt) {
        wait_async0();                   // my staged writes to tile[buf] landed
        __syncthreads();                 // ... and everyone else's
        const int buf = kt & 1;

        // fire-and-forget fill of the next tile (completes before next barrier)
        const int kn = kt + 1;
        if (filler && kn < KT) {
            if (kn * 32 + fc < P) {      // all-or-nothing: fc mult of 8, P mult of 16
                uint32_t lds = (uint32_t)(uintptr_t)(&tile[buf ^ 1][fe]);
                const float* g = xrow + kn * 32;
                async_b128(lds,      g);
                async_b128(lds + 16, g + 4);
            } else {                     // zero-pad ragged tail of K (kn == 24 only)
                v4f z = (v4f){0.f, 0.f, 0.f, 0.f};
                *(v4f*)(&tile[buf ^ 1][fe])     = z;
                *(v4f*)(&tile[buf ^ 1][fe + 4]) = z;
            }
        }

        // A fragments for both M-tiles: ds_load f32, convert to f16 in-register
        const float* ap = &tile[buf][aoff];
        v8f a0l = *(const v8f*)(ap);
        v8f a0h = *(const v8f*)(ap + 8);
        v8f a1l = *(const v8f*)(ap + 16 * 32);
        v8f a1h = *(const v8f*)(ap + 16 * 32 + 8);
        v16f a0f = __builtin_shufflevector(a0l, a0h,
                     0,1,2,3,4,5,6,7,8,9,10,11,12,13,14,15);
        v16f a1f = __builtin_shufflevector(a1l, a1h,
                     0,1,2,3,4,5,6,7,8,9,10,11,12,13,14,15);
        v16h a0 = __builtin_convertvector(a0f, v16h);
        v16h a1 = __builtin_convertvector(a1f, v16h);

        // 7 B fragments, each consumed by two WMMAs (M-tile 0 and 1)
        const _Float16* bp = wbase + (size_t)kt * TILE_HALFS;
        v16h bfrag[7];
#pragma unroll
        for (int f = 0; f < 7; ++f)
            bfrag[f] = *(const v16h*)(bp + (size_t)f * KT * TILE_HALFS);

#pragma unroll
        for (int f = 0; f < 7; ++f) {
            acc0[f] = __builtin_amdgcn_wmma_f32_16x16x32_f16(
                false, a0, false, bfrag[f], (short)0, acc0[f], false, false);
            acc1[f] = __builtin_amdgcn_wmma_f32_16x16x32_f16(
                false, a1, false, bfrag[f], (short)0, acc1[f], false, false);
        }
    }

    // D layout: VGPR r -> M = r + (lane/16)*8 ; N = lane%16 (16x16 f32 C/D)
    const int m0 = b0 + ((lane >> 4) << 3);
    const int nl = lane & 15;
#pragma unroll
    for (int f = 0; f < 7; ++f) {
        int n = (wave * 7 + f) * 16 + nl;
#pragma unroll
        for (int r = 0; r < 8; ++r) {
            out[(size_t)(m0 + r) * P + n]      = acc0[f][r];
            out[(size_t)(m0 + 16 + r) * P + n] = acc1[f][r];
        }
    }
}

// ---------------------------------------------------------------------------
extern "C" void kernel_launch(void* const* d_in, const int* in_sizes, int n_in,
                              void* d_out, int out_size, void* d_ws, size_t ws_size,
                              hipStream_t stream) {
    const float* x      = (const float*)d_in[0]; // (B,1,28,28) f32
    const float* pos2d  = (const float*)d_in[1]; // (28,28,2)   f32
    const float* weight = (const float*)d_in[2]; // (28,28)     f32
    float*       out    = (float*)d_out;         // (B,1,28,28) f32

    _Float16* wswz = (_Float16*)d_ws;            // 49*25*1024 = 1.2 MB scratch

    const int B = in_sizes[0] / P;               // 32768

    {
        const int total = NT * KT * TILE_HALFS;  // 627200
        axs_build_w<<<(total + 255) / 256, 256, 0, stream>>>(pos2d, weight, wswz);
    }
    axs_gemm<<<B / MBLK, 224, 0, stream>>>(x, wswz, out);
}
